// AttentionWithEpinions_8392366096483
// MI455X (gfx1250) — compile-verified, matrix operand order checked by reading, not run on hardware
//
#include <hip/hip_runtime.h>
#include <hip/hip_bf16.h>
#include <stdint.h>

#define D      128
#define TILE_M 64
#define SA     136   // LDS row stride (ushort units) for activation tiles (pad vs bank conflicts)
#define SW     136   // LDS row stride (ushort units) for transposed weight tiles

typedef __attribute__((ext_vector_type(16))) __bf16 v16bf;
typedef __attribute__((ext_vector_type(8)))  float  v8f;

union Frag { v16bf v; unsigned u[8]; };
union Acc  { v8f  v; float   f[8]; };

// ---------- small helpers ----------
__device__ __forceinline__ unsigned short f2bf(float f) {
  unsigned u = __float_as_uint(f);
  unsigned r = ((u >> 16) & 1u) + 0x7FFFu;        // round-to-nearest-even
  return (unsigned short)((u + r) >> 16);
}
__device__ __forceinline__ unsigned pack2(float lo, float hi) {
  return (unsigned)f2bf(lo) | ((unsigned)f2bf(hi) << 16);
}
__device__ __forceinline__ float leaky(float x) { return x > 0.f ? x : 0.01f * x; }

// order-preserving float<->uint encoding for atomicMax-based segment max
__device__ __forceinline__ unsigned enc_ord(float x) {
  unsigned u = __float_as_uint(x);
  return (u & 0x80000000u) ? ~u : (u | 0x80000000u);
}
__device__ __forceinline__ float dec_ord(unsigned u) {
  unsigned b = (u & 0x80000000u) ? (u & 0x7FFFFFFFu) : ~u;
  return __uint_as_float(b);
}

// ---------- LDS staging (blockDim.x == 256) ----------
// W row-major [k][n] (f32) -> bf16 transposed: sWt[n*SW + k]
__device__ __forceinline__ void stage_weight_t(const float* __restrict__ W,
                                               unsigned short* sWt) {
  int t  = threadIdx.x;
  int k  = t >> 1;
  int nb = (t & 1) * 64;
  const float4* Wr = (const float4*)(W + k * D + nb);
#pragma unroll
  for (int q = 0; q < 16; ++q) {
    float4 v = Wr[q];
    int n = nb + q * 4;
    sWt[(n + 0) * SW + k] = f2bf(v.x);
    sWt[(n + 1) * SW + k] = f2bf(v.y);
    sWt[(n + 2) * SW + k] = f2bf(v.z);
    sWt[(n + 3) * SW + k] = f2bf(v.w);
  }
}

// 64 rows of X (f32, row stride D) -> bf16 row-major tile sT[r*SA + col]
__device__ __forceinline__ void stage_rows(const float* __restrict__ X, long row0,
                                           long nrows_max, unsigned short* sT) {
  int t  = threadIdx.x;
  int r  = t >> 2;
  int cb = (t & 3) * 32;
  long row = row0 + r;
  if (row >= nrows_max) row = nrows_max - 1;   // clamp (tail-safe)
  const float4* p = (const float4*)(X + row * D + cb);
#pragma unroll
  for (int q = 0; q < 8; ++q) {
    float4 v = p[q];
    uint2 d;
    d.x = pack2(v.x, v.y);
    d.y = pack2(v.z, v.w);
    *(uint2*)&sT[r * SA + cb + q * 4] = d;
  }
}

// ---------- WMMA fragment loads (layouts per ISA 7.12.2, 16-bit 16x32 A / 32x16 B) ----------
__device__ __forceinline__ Frag load_a(const unsigned short* sT, int m0, int k0, int lane) {
  Frag a;
  int m  = m0 + (lane & 15);
  int kb = k0 + ((lane & 16) ? 8 : 0);
  const unsigned short* p = sT + m * SA;
#pragma unroll
  for (int i = 0; i < 8; ++i) {
    int k = kb + ((i & 4) ? 16 : 0) + ((i & 3) << 1);
    a.u[i] = *(const unsigned*)(p + k);   // pair (k,k+1) contiguous: one b32 LDS read
  }
  return a;
}
__device__ __forceinline__ Frag load_b(const unsigned short* sWt, int n0, int k0, int lane) {
  Frag b;
  int n  = n0 + (lane & 15);
  int kb = k0 + ((lane & 16) ? 8 : 0);
  const unsigned short* p = sWt + n * SW;
#pragma unroll
  for (int i = 0; i < 8; ++i) {
    int k = kb + ((i & 4) ? 16 : 0) + ((i & 3) << 1);
    b.u[i] = *(const unsigned*)(p + k);
  }
  return b;
}

// one wave computes 16(M) x 64(N) of a [64x128]@[128x128] product
__device__ __forceinline__ void gemm_64x128(const unsigned short* sT, const unsigned short* sWt,
                                            Acc acc[4], int m0, int n0, int lane) {
#pragma unroll
  for (int ks = 0; ks < 4; ++ks) {
    Frag a = load_a(sT, m0, ks * 32, lane);
#pragma unroll
    for (int t = 0; t < 4; ++t) {
      Frag b = load_b(sWt, n0 + t * 16, ks * 32, lane);
      acc[t].v = __builtin_amdgcn_wmma_f32_16x16x32_bf16(
          false, a.v, false, b.v, (short)0, acc[t].v, false, false);
    }
  }
}

// ---------- kernels ----------
__global__ void init_seg_kernel(unsigned* seg_max_u, float* seg_sum, int n) {
  int i = blockIdx.x * blockDim.x + threadIdx.x;
  if (i < n) { seg_max_u[i] = 0u; seg_sum[i] = 0.f; }
}

// r_ft = dst_feat @ W_dst + b_dst   (f32 out, bf16 WMMA inside)
__global__ __launch_bounds__(256) void node_linear_kernel(
    const float* __restrict__ X, const float* __restrict__ W,
    const float* __restrict__ bias, float* __restrict__ Y, int nrows) {
  __shared__ __align__(16) unsigned short sWt[D * SW];
  __shared__ __align__(16) unsigned short sT[TILE_M * SA];
  __shared__ float sB[D];
  stage_weight_t(W, sWt);
  long row0 = (long)blockIdx.x * TILE_M;
  stage_rows(X, row0, nrows, sT);
  if (threadIdx.x < D) sB[threadIdx.x] = bias[threadIdx.x];
  __syncthreads();

  int w = threadIdx.x >> 5, lane = threadIdx.x & 31;
  int m0 = (w >> 1) * 16, n0 = (w & 1) * 64;
  int mh = (lane & 16) ? 8 : 0, nl = lane & 15;

  Acc acc[4];
#pragma unroll
  for (int t = 0; t < 4; ++t)
#pragma unroll
    for (int i = 0; i < 8; ++i) acc[t].f[i] = sB[n0 + t * 16 + nl];
  gemm_64x128(sT, sWt, acc, m0, n0, lane);

#pragma unroll
  for (int t = 0; t < 4; ++t)
#pragma unroll
    for (int i = 0; i < 8; ++i) {
      long row = row0 + m0 + i + mh;
      if (row < nrows) Y[row * D + n0 + t * 16 + nl] = acc[t].f[i];
    }
}

// fused per-edge: (src@Wsrc+bsrc + r_ft[dst]) -> leaky -> @W1+b1 -> leaky -> dot W2 + b2
// writes logit to d_out[e] and atomicMax into seg_max
__global__ __launch_bounds__(256) void edge_fused_kernel(
    const float* __restrict__ src_feat, const float* __restrict__ Wsrc,
    const float* __restrict__ bsrc, const float* __restrict__ W1,
    const float* __restrict__ b1, const float* __restrict__ W2,
    const float* __restrict__ b2, const float* __restrict__ r_ft,
    const int* __restrict__ edge_dst, float* __restrict__ logits_out,
    unsigned* __restrict__ seg_max_u, long nedges) {
  __shared__ __align__(16) unsigned short sWa[D * SW];
  __shared__ __align__(16) unsigned short sWb[D * SW];
  __shared__ __align__(16) unsigned short sA_t[TILE_M * SA];
  __shared__ __align__(16) unsigned short sS_t[TILE_M * SA];
  __shared__ float sB0[D], sB1[D], sW2v[D];
  __shared__ int   sDst[TILE_M];
  __shared__ float sLogit[TILE_M];

  stage_weight_t(Wsrc, sWa);
  stage_weight_t(W1, sWb);
  long e0 = (long)blockIdx.x * TILE_M;
  stage_rows(src_feat, e0, nedges, sA_t);
  int t = threadIdx.x;
  if (t < D) { sB0[t] = bsrc[t]; sB1[t] = b1[t]; sW2v[t] = W2[t]; }
  if (t < TILE_M) {
    long e = e0 + t; if (e >= nedges) e = nedges - 1;
    sDst[t]   = edge_dst[e];
    sLogit[t] = b2[0];
  }
  __syncthreads();
  if (t < TILE_M) __builtin_prefetch(&r_ft[(long)sDst[t] * D], 0, 0); // warm L2 gather rows

  int w = t >> 5, lane = t & 31;
  int m0 = (w >> 1) * 16, n0 = (w & 1) * 64;
  int mh = (lane & 16) ? 8 : 0, nl = lane & 15;

  // GEMM1, C preloaded with b_src[n] + gathered r_ft[dst][n]
  Acc acc[4];
#pragma unroll
  for (int tt = 0; tt < 4; ++tt) {
    int n = n0 + tt * 16 + nl;
#pragma unroll
    for (int i = 0; i < 8; ++i) {
      int mi = m0 + i + mh;
      acc[tt].f[i] = sB0[n] + r_ft[(long)sDst[mi] * D + n];
    }
  }
  gemm_64x128(sA_t, sWa, acc, m0, n0, lane);

  // leaky-ReLU, stash score tile as bf16 for GEMM2's A operand
#pragma unroll
  for (int tt = 0; tt < 4; ++tt) {
    int n = n0 + tt * 16 + nl;
#pragma unroll
    for (int i = 0; i < 8; ++i)
      sS_t[(m0 + i + mh) * SA + n] = f2bf(leaky(acc[tt].f[i]));
  }
  __syncthreads();

  // GEMM2, C preloaded with b1[n]
  Acc h[4];
#pragma unroll
  for (int tt = 0; tt < 4; ++tt)
#pragma unroll
    for (int i = 0; i < 8; ++i) h[tt].f[i] = sB1[n0 + tt * 16 + nl];
  gemm_64x128(sS_t, sWb, h, m0, n0, lane);

  // GEMV with W2: leaky(h)*w2, reduce over the 64 n-columns this wave owns
#pragma unroll
  for (int i = 0; i < 8; ++i) {
    float x = 0.f;
#pragma unroll
    for (int tt = 0; tt < 4; ++tt)
      x += leaky(h[tt].f[i]) * sW2v[n0 + tt * 16 + nl];
    x += __shfl_xor(x, 8, 16);
    x += __shfl_xor(x, 4, 16);
    x += __shfl_xor(x, 2, 16);
    x += __shfl_xor(x, 1, 16);
    if (nl == 0) atomicAdd(&sLogit[m0 + i + mh], x);   // ds_add_f32
  }
  __syncthreads();

  if (t < TILE_M) {
    long e = e0 + t;
    if (e < nedges) {
      float lg = sLogit[t];
      logits_out[e] = lg;
      atomicMax(&seg_max_u[sDst[t]], enc_ord(lg));     // global_atomic_max_u32
    }
  }
}

__global__ void softmax_p1_kernel(float* __restrict__ io, const int* __restrict__ edge_dst,
                                  const unsigned* __restrict__ seg_max_u,
                                  float* __restrict__ seg_sum, long E) {
  long e = (long)blockIdx.x * blockDim.x + threadIdx.x;
  if (e < E) {
    int d = edge_dst[e];
    float ex = __expf(io[e] - dec_ord(seg_max_u[d]));
    io[e] = ex;
    atomicAdd(&seg_sum[d], ex);                        // global_atomic_add_f32
  }
}

__global__ void softmax_p2_kernel(float* __restrict__ io, const int* __restrict__ edge_dst,
                                  const float* __restrict__ seg_sum, long E) {
  long e = (long)blockIdx.x * blockDim.x + threadIdx.x;
  if (e < E) io[e] = io[e] / seg_sum[edge_dst[e]];
}

// ---------- launcher ----------
extern "C" void kernel_launch(void* const* d_in, const int* in_sizes, int n_in,
                              void* d_out, int out_size, void* d_ws, size_t ws_size,
                              hipStream_t stream) {
  const float* src_feat = (const float*)d_in[0];
  const float* dst_feat = (const float*)d_in[1];
  const int*   edge_dst = (const int*)d_in[2];
  const float* W_src    = (const float*)d_in[3];
  const float* b_src    = (const float*)d_in[4];
  const float* W_dst    = (const float*)d_in[5];
  const float* b_dst    = (const float*)d_in[6];
  const float* W1       = (const float*)d_in[7];
  const float* b1       = (const float*)d_in[8];
  const float* W2       = (const float*)d_in[9];
  const float* b2       = (const float*)d_in[10];

  const long E = in_sizes[2];       // n_edges
  const int  N = in_sizes[1] / D;   // n_nodes

  char* wsp = (char*)d_ws;
  float*    r_ft      = (float*)wsp;     wsp += (size_t)N * D * sizeof(float);
  unsigned* seg_max_u = (unsigned*)wsp;  wsp += (size_t)N * sizeof(unsigned);
  float*    seg_sum   = (float*)wsp;

  float* out = (float*)d_out;

  init_seg_kernel<<<(N + 255) / 256, 256, 0, stream>>>(seg_max_u, seg_sum, N);
  node_linear_kernel<<<(N + TILE_M - 1) / TILE_M, 256, 0, stream>>>(
      dst_feat, W_dst, b_dst, r_ft, N);
  edge_fused_kernel<<<(int)((E + TILE_M - 1) / TILE_M), 256, 0, stream>>>(
      src_feat, W_src, b_src, W1, b1, W2, b2, r_ft, edge_dst, out, seg_max_u, E);
  softmax_p1_kernel<<<(int)((E + 255) / 256), 256, 0, stream>>>(
      out, edge_dst, seg_max_u, seg_sum, E);
  softmax_p2_kernel<<<(int)((E + 255) / 256), 256, 0, stream>>>(
      out, edge_dst, seg_sum, E);
}
